// GATNet_simple_44281112822533
// MI455X (gfx1250) — compile-verified, matrix operand order checked by reading, not run on hardware
//
#include <hip/hip_runtime.h>
#include <hip/hip_bf16.h>

typedef __attribute__((ext_vector_type(16))) __bf16 v16bf;
typedef __attribute__((ext_vector_type(8)))  __bf16 v8bf;
typedef __attribute__((ext_vector_type(8)))  float  v8f;

#define NEG_SLOPE 0.2f
#define GAT_EPS 1e-16f

// ---------------- helpers ----------------

static __device__ __forceinline__ __bf16 f2bf(float f) {
  // round-to-nearest-even float -> bf16 (bit-level, avoids relying on cast lowering)
  unsigned u = __float_as_uint(f);
  unsigned r = (u + 0x7FFFu + ((u >> 16) & 1u)) >> 16;
  unsigned short h = (unsigned short)r;
  __bf16 out;
  __builtin_memcpy(&out, &h, sizeof(out));
  return out;
}

// order-preserving float <-> uint mapping for atomicMax on floats
static __device__ __forceinline__ unsigned fflip(float f) {
  unsigned u = __float_as_uint(f);
  return (u & 0x80000000u) ? ~u : (u | 0x80000000u);
}
static __device__ __forceinline__ float funflip(unsigned u) {
  return (u & 0x80000000u) ? __uint_as_float(u & 0x7FFFFFFFu) : __uint_as_float(~u);
}

static __device__ __forceinline__ float lrelu(float e) {
  return e > 0.f ? e : NEG_SLOPE * e;
}

// ---- CDNA5 async copy: global -> LDS, 16 bytes per lane, tracked by ASYNCcnt ----
// The LDS destination is passed as a generic pointer; its low 32 bits ARE the LDS
// byte offset (generic LDS address = {SHARED_BASE aperture, offset[31:0]}, ISA 10.2).
// Using ptrtoint of the real pointer (not a synthesized integer) makes the LDS
// object escape, so the "memory" clobber keeps subsequent ds_loads of it alive.
static __device__ __forceinline__ void async_copy_b128(void* lds, const void* gaddr) {
  unsigned lds_off = (unsigned)(unsigned long long)lds;
  asm volatile("global_load_async_to_lds_b128 %0, %1, off"
               :: "v"(lds_off), "v"((unsigned long long)gaddr)
               : "memory");
}

static __device__ __forceinline__ void wait_async0() {
#if __has_builtin(__builtin_amdgcn_s_wait_asynccnt)
  __builtin_amdgcn_s_wait_asynccnt(0);
#else
  asm volatile("s_wait_asynccnt 0" ::: "memory");
#endif
}

// ---------------- conversion / packing ----------------

// x [N,16] f32 -> xb [N,32] bf16, upper 16 columns zero (pads layer-1 K to 32)
__global__ void k_x_pad_bf16(const float* __restrict__ in, __bf16* __restrict__ out,
                             int N) {
  int i = blockIdx.x * blockDim.x + threadIdx.x;
  if (i >= N * 32) return;
  int n = i >> 5, j = i & 31;
  out[i] = (j < 16) ? f2bf(in[n * 16 + j]) : f2bf(0.f);
}

// Pack row-major B[K,Nc] (f32) into per-lane WMMA bf16 fragments:
// Bp[((kt*ntilesN + tn)*32 + lane)*16 + e], lane 0-15 -> col, halfsel = lane>>4,
// e<8 -> k = kt*32 + halfsel*8 + e ; e>=8 -> k = kt*32 + halfsel*8 + 16 + (e-8)
__global__ void k_pack_b(const float* __restrict__ B, __bf16* __restrict__ Bp,
                         int K, int Nc) {
  int ktiles = (K + 31) >> 5;
  int ntiles = Nc >> 4;
  int total = ktiles * ntiles * 32 * 16;
  int idx = blockIdx.x * blockDim.x + threadIdx.x;
  if (idx >= total) return;
  int e    = idx & 15;
  int lane = (idx >> 4) & 31;
  int tile = idx >> 9;
  int tn = tile % ntiles;
  int kt = tile / ntiles;
  int n  = tn * 16 + (lane & 15);
  int kb = kt * 32 + (lane >> 4) * 8;
  int k  = kb + (e < 8 ? e : 8 + e);   // e>=8: kb+16+(e-8)
  float v = (k < K) ? B[(long long)k * Nc + n] : 0.0f;
  Bp[idx] = f2bf(v);
}

// ---------------- WMMA GEMM: C[M,Nc](f32) = A[M,K](bf16) x Bpacked ----------------
// Requirements: M % 16 == 0, K % 32 == 0 (zero-padded), Nc % 64 == 0.
// Block = 8 waves = 8 M-tiles x one 64-col strip. All waves share the same packed-B
// k-tile (4 fragments = 4KB), staged into LDS with double-buffered async copies.
__global__ void __launch_bounds__(256) k_gemm_wmma(
    const __bf16* __restrict__ A, const __bf16* __restrict__ Bp,
    float* __restrict__ C, int M, int K, int Nc) {
  __shared__ __bf16 smem[2 * 2048];     // 2 buffers x (4 tiles x 32 lanes x 16 elems)

  int ntn    = Nc >> 4;                 // 16-col tiles
  int nstrip = ntn >> 2;                // 64-col strips
  int ntm    = M >> 4;
  int tid  = threadIdx.x;
  int wv   = tid >> 5;
  int lane = tid & 31;
  int bm   = blockIdx.x / nstrip;
  int tn0  = (blockIdx.x % nstrip) << 2;
  int tm   = bm * 8 + wv;
  bool valid = (tm < ntm);              // wave-uniform
  int halfsel = lane >> 4;
  const __bf16* __restrict__ arowp =
      A + (long long)((valid ? tm : 0) * 16 + (lane & 15)) * K + halfsel * 8;
  int ktiles = K >> 5;

  // stage k-tile 0 (4KB) into LDS buffer 0: one b128 async copy per thread
  const __bf16* bbase = Bp + (long long)tn0 * 512;
  async_copy_b128(smem + tid * 8, bbase + tid * 8);
  wait_async0();
  __syncthreads();

  v8f acc0 = {}, acc1 = {}, acc2 = {}, acc3 = {};
  for (int kt = 0; kt < ktiles; ++kt) {
    int cur = kt & 1;
    if (kt + 1 < ktiles) {                        // prefetch next k-tile
      async_copy_b128(smem + (cur ^ 1) * 2048 + tid * 8,
                      bbase + (long long)(kt + 1) * ntn * 512 + tid * 8);
    }
    if (valid) {
      v8bf r0 = *(const v8bf*)(arowp + kt * 32);        // K run [kb, kb+8)
      v8bf r1 = *(const v8bf*)(arowp + kt * 32 + 16);   // K run [kb+16, kb+24)
      v16bf afrag = __builtin_shufflevector(r0, r1,
          0, 1, 2, 3, 4, 5, 6, 7, 8, 9, 10, 11, 12, 13, 14, 15);
      const __bf16* sb = smem + cur * 2048 + lane * 16;
      v16bf b0 = *(const v16bf*)(sb);
      v16bf b1 = *(const v16bf*)(sb + 512);
      v16bf b2 = *(const v16bf*)(sb + 1024);
      v16bf b3 = *(const v16bf*)(sb + 1536);
      acc0 = __builtin_amdgcn_wmma_f32_16x16x32_bf16(false, afrag, false, b0, (short)0, acc0, false, false);
      acc1 = __builtin_amdgcn_wmma_f32_16x16x32_bf16(false, afrag, false, b1, (short)0, acc1, false, false);
      acc2 = __builtin_amdgcn_wmma_f32_16x16x32_bf16(false, afrag, false, b2, (short)0, acc2, false, false);
      acc3 = __builtin_amdgcn_wmma_f32_16x16x32_bf16(false, afrag, false, b3, (short)0, acc3, false, false);
    }
    wait_async0();      // prefetch complete (harmless when none outstanding)
    __syncthreads();    // publish LDS buffer for next iteration
  }

  if (valid) {
    // C/D layout: vgpr v, lane L -> row = tm*16 + v + 8*(L>=16), col = (L&15)
    int rbase = tm * 16 + halfsel * 8;
    float* cp = C + (long long)rbase * Nc + tn0 * 16 + (lane & 15);
#pragma unroll
    for (int v = 0; v < 8; ++v) {
      cp[(long long)v * Nc +  0] = acc0[v];
      cp[(long long)v * Nc + 16] = acc1[v];
      cp[(long long)v * Nc + 32] = acc2[v];
      cp[(long long)v * Nc + 48] = acc3[v];
    }
  }
}

// ---------------- GAT edge phase ----------------

// per-node attention scores: as[n,h] = sum_c h[n,h,c]*a_src[h,c]; same for ad
__global__ void k_node_scores(const float* __restrict__ h, const float* __restrict__ a_src,
                              const float* __restrict__ a_dst, float* __restrict__ as_,
                              float* __restrict__ ad_, int N, int H, int C) {
  int idx = blockIdx.x * blockDim.x + threadIdx.x;
  if (idx >= N * H) return;
  int n = idx / H, hh = idx % H;
  const float* hp = h + ((long long)n * H + hh) * C;
  const float* sp = a_src + hh * C;
  const float* dp = a_dst + hh * C;
  float s = 0.f, d = 0.f;
  for (int c = 0; c < C; ++c) { float v = hp[c]; s += v * sp[c]; d += v * dp[c]; }
  as_[idx] = s;
  ad_[idx] = d;
}

__global__ void k_edge_max(const int* __restrict__ ei, int E, int Etot,
                           const float* __restrict__ as_, const float* __restrict__ ad_,
                           unsigned* __restrict__ m, int H) {
  long long idx = (long long)blockIdx.x * blockDim.x + threadIdx.x;
  if (idx >= (long long)Etot * H) return;
  int eid = (int)(idx / H), hh = (int)(idx % H);
  int s, d;
  if (eid < E) { s = ei[eid]; d = ei[E + eid]; } else { s = d = eid - E; }
  float e = lrelu(as_[s * H + hh] + ad_[d * H + hh]);
  atomicMax(&m[d * H + hh], fflip(e));
}

__global__ void k_edge_expsum(const int* __restrict__ ei, int E, int Etot,
                              const float* __restrict__ as_, const float* __restrict__ ad_,
                              const unsigned* __restrict__ m, float* __restrict__ den, int H) {
  long long idx = (long long)blockIdx.x * blockDim.x + threadIdx.x;
  if (idx >= (long long)Etot * H) return;
  int eid = (int)(idx / H), hh = (int)(idx % H);
  int s, d;
  if (eid < E) { s = ei[eid]; d = ei[E + eid]; } else { s = d = eid - E; }
  float e = lrelu(as_[s * H + hh] + ad_[d * H + hh]);
  float ex = __expf(e - funflip(m[d * H + hh]));
  atomicAdd(&den[d * H + hh], ex);
}

__global__ void k_edge_alpha(const int* __restrict__ ei, int E, int Etot,
                             const float* __restrict__ as_, const float* __restrict__ ad_,
                             const unsigned* __restrict__ m, const float* __restrict__ den,
                             float* __restrict__ alpha, int H) {
  long long idx = (long long)blockIdx.x * blockDim.x + threadIdx.x;
  if (idx >= (long long)Etot * H) return;
  int eid = (int)(idx / H), hh = (int)(idx % H);
  int s, d;
  if (eid < E) { s = ei[eid]; d = ei[E + eid]; } else { s = d = eid - E; }
  float e = lrelu(as_[s * H + hh] + ad_[d * H + hh]);
  float ex = __expf(e - funflip(m[d * H + hh]));
  alpha[idx] = ex / (den[d * H + hh] + GAT_EPS);
}

// scatter: xout[dst,h,c] += h[src,h,c]*alpha ; 4 channels per thread (float4 gather)
__global__ void k_edge_scatter(const int* __restrict__ ei, int E, int Etot,
                               const float* __restrict__ h, const float* __restrict__ alpha,
                               float* __restrict__ xout, int H, int C) {
  long long idx = (long long)blockIdx.x * blockDim.x + threadIdx.x;
  int cpe = C >> 2;
  long long total = (long long)Etot * H * cpe;
  if (idx >= total) return;
  int chunk = (int)(idx % cpe);
  long long rest = idx / cpe;
  int hh  = (int)(rest % H);
  int eid = (int)(rest / H);
  int s, d;
  if (eid < E) { s = ei[eid]; d = ei[E + eid]; } else { s = d = eid - E; }
  float a = alpha[(long long)eid * H + hh];
  const float4 hv = *(const float4*)(h + ((long long)s * H + hh) * C + chunk * 4);
  float* op = xout + ((long long)d * H + hh) * C + chunk * 4;
  atomicAdd(op + 0, hv.x * a);
  atomicAdd(op + 1, hv.y * a);
  atomicAdd(op + 2, hv.z * a);
  atomicAdd(op + 3, hv.w * a);
}

// x = relu(x + bias); optional bf16 mirror for the next layer's GEMM input
__global__ void k_bias_relu(float* __restrict__ x, __bf16* __restrict__ xb,
                            const float* __restrict__ bias, long long n, int HC) {
  long long idx = (long long)blockIdx.x * blockDim.x + threadIdx.x;
  if (idx >= n) return;
  float v = x[idx] + bias[(int)(idx % HC)];
  v = v > 0.f ? v : 0.f;
  x[idx] = v;
  if (xb) xb[idx] = f2bf(v);
}

// final head: out[n] = sigmoid(concat(x1,x2,x3)[n] . Wf + bf), one wave per node
__global__ void k_final(const float* __restrict__ x1, const float* __restrict__ x2,
                        const float* __restrict__ x3, const float* __restrict__ Wf,
                        const float* __restrict__ bf, float* __restrict__ out, int N) {
  int gw = (int)((blockIdx.x * (long long)blockDim.x + threadIdx.x) >> 5);
  int lane = threadIdx.x & 31;
  if (gw >= N) return;
  float acc = 0.f;
  for (int j = lane; j < 256; j += 32) acc += x1[(long long)gw * 256 + j] * Wf[j];
  for (int j = lane; j < 256; j += 32) acc += x2[(long long)gw * 256 + j] * Wf[256 + j];
  for (int j = lane; j < 768; j += 32) acc += x3[(long long)gw * 768 + j] * Wf[512 + j];
#pragma unroll
  for (int off = 16; off; off >>= 1) acc += __shfl_xor(acc, off, 32);
  if (lane == 0) out[gw] = 1.f / (1.f + __expf(-(acc + bf[0])));
}

// ---------------- host driver ----------------

static inline int cdiv_i(long long a, int b) { return (int)((a + b - 1) / b); }

struct EdgeCtx {
  const int* ei; int E, Etot;
  float* as_; float* ad_; unsigned* m; float* den; float* alpha;
};

static void run_gat_layer(hipStream_t stream, int N, const EdgeCtx& ec,
                          const __bf16* Ain, int K, int H, int C,
                          const __bf16* Wp, const float* a_src, const float* a_dst,
                          const float* bias, float* hbuf, float* xout, __bf16* xoutb) {
  const int HC = H * C;
  // 1) projection GEMM via WMMA (8 M-tiles x 64-col strip per block, LDS-staged B)
  {
    int nstrip = HC >> 6;
    int nbm = cdiv_i(N >> 4, 8);
    k_gemm_wmma<<<nbm * nstrip, 256, 0, stream>>>(Ain, Wp, hbuf, N, K, HC);
  }
  // 2) clear accumulators / softmax state
  hipMemsetAsync(xout, 0, (size_t)N * HC * sizeof(float), stream);
  hipMemsetAsync(ec.m, 0, (size_t)N * H * sizeof(unsigned), stream);   // fflip sentinel < any real score
  hipMemsetAsync(ec.den, 0, (size_t)N * H * sizeof(float), stream);
  // 3) node scores
  k_node_scores<<<cdiv_i((long long)N * H, 256), 256, 0, stream>>>(
      hbuf, a_src, a_dst, ec.as_, ec.ad_, N, H, C);
  // 4-6) segment softmax over edges
  long long eh = (long long)ec.Etot * H;
  k_edge_max<<<cdiv_i(eh, 256), 256, 0, stream>>>(ec.ei, ec.E, ec.Etot, ec.as_, ec.ad_, ec.m, H);
  k_edge_expsum<<<cdiv_i(eh, 256), 256, 0, stream>>>(ec.ei, ec.E, ec.Etot, ec.as_, ec.ad_, ec.m, ec.den, H);
  k_edge_alpha<<<cdiv_i(eh, 256), 256, 0, stream>>>(ec.ei, ec.E, ec.Etot, ec.as_, ec.ad_, ec.m, ec.den, ec.alpha, H);
  // 7) weighted message scatter
  k_edge_scatter<<<cdiv_i(eh * (C >> 2), 256), 256, 0, stream>>>(
      ec.ei, ec.E, ec.Etot, hbuf, ec.alpha, xout, H, C);
  // 8) bias + relu (+ bf16 mirror)
  k_bias_relu<<<cdiv_i((long long)N * HC, 256), 256, 0, stream>>>(
      xout, xoutb, bias, (long long)N * HC, HC);
}

extern "C" void kernel_launch(void* const* d_in, const int* in_sizes, int n_in,
                              void* d_out, int out_size, void* d_ws, size_t ws_size,
                              hipStream_t stream) {
  const float* x   = (const float*)d_in[0];
  const int*   ei  = (const int*)d_in[1];
  const float* W1  = (const float*)d_in[3];
  const float* as1 = (const float*)d_in[4];
  const float* ad1 = (const float*)d_in[5];
  const float* b1  = (const float*)d_in[6];
  const float* W2  = (const float*)d_in[7];
  const float* as2 = (const float*)d_in[8];
  const float* ad2 = (const float*)d_in[9];
  const float* b2  = (const float*)d_in[10];
  const float* W3  = (const float*)d_in[11];
  const float* as3 = (const float*)d_in[12];
  const float* ad3 = (const float*)d_in[13];
  const float* b3  = (const float*)d_in[14];
  const float* Wf  = (const float*)d_in[15];
  const float* bf  = (const float*)d_in[16];

  const int N = in_sizes[0] / 16;      // 50000 (divisible by 16)
  const int E = in_sizes[1] / 2;       // 500000
  const int Etot = E + N;              // + self loops
  const int Hmax = 12;

  // bump allocator over d_ws
  char* p = (char*)d_ws;
  auto alloc = [&](size_t bytes) -> void* {
    void* r = (void*)p;
    p += (bytes + 255) & ~(size_t)255;
    return r;
  };

  __bf16* xb   = (__bf16*)alloc((size_t)N * 32 * 2);          // K padded 16 -> 32
  __bf16* W1p  = (__bf16*)alloc((size_t)1 * 16 * 512 * 2);   // ktiles(16->32)=1, ntiles=16
  __bf16* W2p  = (__bf16*)alloc((size_t)8 * 16 * 512 * 2);   // ktiles(256)=8
  __bf16* W3p  = (__bf16*)alloc((size_t)8 * 48 * 512 * 2);
  float*  hbuf = (float*)alloc((size_t)N * 768 * 4);          // shared across layers
  float*  x1   = (float*)alloc((size_t)N * 256 * 4);
  __bf16* x1b  = (__bf16*)alloc((size_t)N * 256 * 2);
  float*  x2   = (float*)alloc((size_t)N * 256 * 4);
  __bf16* x2b  = (__bf16*)alloc((size_t)N * 256 * 2);
  float*  x3   = (float*)alloc((size_t)N * 768 * 4);
  float*  as_  = (float*)alloc((size_t)N * Hmax * 4);
  float*  ad_  = (float*)alloc((size_t)N * Hmax * 4);
  unsigned* m  = (unsigned*)alloc((size_t)N * Hmax * 4);
  float*  den  = (float*)alloc((size_t)N * Hmax * 4);
  float*  alpha = (float*)alloc((size_t)Etot * Hmax * 4);

  // input conversion + weight packing (tiny)
  k_x_pad_bf16<<<cdiv_i((long long)N * 32, 256), 256, 0, stream>>>(x, xb, N);
  k_pack_b<<<cdiv_i(1 * 16 * 512, 256), 256, 0, stream>>>(W1, W1p, 16, 256);
  k_pack_b<<<cdiv_i(8 * 16 * 512, 256), 256, 0, stream>>>(W2, W2p, 256, 256);
  k_pack_b<<<cdiv_i(8 * 48 * 512, 256), 256, 0, stream>>>(W3, W3p, 256, 768);

  EdgeCtx ec{ei, E, Etot, as_, ad_, m, den, alpha};

  run_gat_layer(stream, N, ec, xb,  32,  8, 32, W1p, as1, ad1, b1, hbuf, x1, x1b);
  run_gat_layer(stream, N, ec, x1b, 256, 8, 32, W2p, as2, ad2, b2, hbuf, x2, x2b);
  run_gat_layer(stream, N, ec, x2b, 256, 12, 64, W3p, as3, ad3, b3, hbuf, x3, nullptr);

  k_final<<<cdiv_i((long long)N * 32, 256), 256, 0, stream>>>(
      x1, x2, x3, Wf, bf, (float*)d_out, N);
}
